// PointNet2Generator_52982716563597
// MI455X (gfx1250) — compile-verified, matrix operand order checked by reading, not run on hardware
//
#include <hip/hip_runtime.h>
#include <cstdint>
#include <cstddef>

typedef unsigned short u16;
typedef unsigned int   u32;

typedef __attribute__((ext_vector_type(16))) __bf16          v16bf;
typedef __attribute__((ext_vector_type(8)))  unsigned int    v8u;
typedef __attribute__((ext_vector_type(4)))  unsigned int    v4u;
typedef __attribute__((ext_vector_type(8)))  float           v8f;

// ---------------------------------------------------------------- helpers
__device__ __forceinline__ u16 f2bf(float f) {
    u32 u = __float_as_uint(f);
    u32 r = u + 0x7FFFu + ((u >> 16) & 1u);   // round-to-nearest-even
    return (u16)(r >> 16);
}
__device__ __forceinline__ float bf2f(u16 h) {
    return __uint_as_float(((u32)h) << 16);
}
__device__ __forceinline__ float silu_f(float x) {
    return x / (1.0f + __expf(-x));
}

// CDNA5 async global->LDS DMA (tracked by ASYNCcnt, no VGPR staging).
// dst-first VFLAT syntax; lds_addr = low 32 bits of generic shared pointer
// (flat->LDS aperture mapping truncates to addr[31:0]).
__device__ __forceinline__ void async_g2l_b128(u32 lds_addr, const u16* gptr) {
    asm volatile("global_load_async_to_lds_b128 %0, %1, off"
                 :: "v"(lds_addr), "v"(gptr)
                 : "memory");
}
__device__ __forceinline__ void wait_async0() {
    asm volatile("s_wait_asynccnt 0x0" ::: "memory");
}

// ---------------------------------------------------------------- pos0 = data^T
__global__ void k_pos0(const float* __restrict__ data, float* __restrict__ pos0, int N) {
    int b = blockIdx.y;
    int i = blockIdx.x * blockDim.x + threadIdx.x;
    if (i >= N) return;
    for (int c = 0; c < 3; c++)
        pos0[((size_t)b * N + i) * 3 + c] = data[(size_t)b * 3 * N + (size_t)c * N + i];
}

// ---------------------------------------------------------------- farthest point sampling
// one block per batch, sequential scan with parallel argmax reduction
__global__ __launch_bounds__(1024) void k_fps(const float* __restrict__ pos, int n, int m,
                                              int* __restrict__ idx) {
    int b = blockIdx.x;
    pos += (size_t)b * n * 3;
    idx += (size_t)b * m;

    __shared__ float dmin[4096];
    __shared__ float rv[1024];
    __shared__ int   ri[1024];
    __shared__ int   sLast;

    int tid = threadIdx.x;
    for (int i = tid; i < n; i += 1024) dmin[i] = 3.4e38f;
    if (tid == 0) sLast = 0;
    __syncthreads();

    for (int j = 0; j < m; j++) {
        int last = sLast;
        if (tid == 0) idx[j] = last;
        float lx = pos[last * 3 + 0];
        float ly = pos[last * 3 + 1];
        float lz = pos[last * 3 + 2];
        float best = -1.0f;
        int   bidx = 0x7fffffff;
        for (int i = tid; i < n; i += 1024) {
            float dx = pos[i * 3 + 0] - lx;
            float dy = pos[i * 3 + 1] - ly;
            float dz = pos[i * 3 + 2] - lz;
            float d  = dx * dx + dy * dy + dz * dz;
            float nd = fminf(dmin[i], d);
            dmin[i] = nd;
            if (nd > best || (nd == best && i < bidx)) { best = nd; bidx = i; }
        }
        rv[tid] = best; ri[tid] = bidx;
        __syncthreads();
        for (int s = 512; s > 0; s >>= 1) {
            if (tid < s) {
                float ov = rv[tid + s]; int oi = ri[tid + s];
                if (ov > rv[tid] || (ov == rv[tid] && oi < ri[tid])) { rv[tid] = ov; ri[tid] = oi; }
            }
            __syncthreads();
        }
        if (tid == 0) sLast = ri[0];
        __syncthreads();
    }
}

// ---------------------------------------------------------------- gather positions
__global__ void k_gather_pos(const float* __restrict__ pos, const int* __restrict__ idx,
                             float* __restrict__ posn, int n, int m) {
    int b = blockIdx.y;
    int j = blockIdx.x * blockDim.x + threadIdx.x;
    if (j >= m) return;
    int s = idx[(size_t)b * m + j];
    for (int c = 0; c < 3; c++)
        posn[((size_t)b * m + j) * 3 + c] = pos[((size_t)b * n + s) * 3 + c];
}

// ---------------------------------------------------------------- ball query (first 32 in-radius, by index)
__global__ void k_ball(const float* __restrict__ pos, const float* __restrict__ q,
                       int n, int m, float r2, int* __restrict__ nb, int* __restrict__ cnt) {
    int b = blockIdx.y;
    int j = blockIdx.x * blockDim.x + threadIdx.x;
    if (j >= m) return;
    const float* P = pos + (size_t)b * n * 3;
    float qx = q[((size_t)b * m + j) * 3 + 0];
    float qy = q[((size_t)b * m + j) * 3 + 1];
    float qz = q[((size_t)b * m + j) * 3 + 2];
    int c = 0;
    int* nbp = nb + ((size_t)b * m + j) * 32;
    for (int i = 0; i < n && c < 32; i++) {
        float dx = qx - P[i * 3 + 0];
        float dy = qy - P[i * 3 + 1];
        float dz = qz - P[i * 3 + 2];
        float d2 = dx * dx + dy * dy + dz * dz;
        if (d2 <= r2) { nbp[c] = i; c++; }
    }
    cnt[(size_t)b * m + j] = c;
}

// ---------------------------------------------------------------- 3-NN with inverse-sq-dist weights
__global__ void k_knn(const float* __restrict__ pc, const float* __restrict__ pf,
                      int nc, int nf, int* __restrict__ ki, float* __restrict__ kw) {
    int b = blockIdx.y;
    int i = blockIdx.x * blockDim.x + threadIdx.x;
    if (i >= nf) return;
    const float* PC = pc + (size_t)b * nc * 3;
    float fx = pf[((size_t)b * nf + i) * 3 + 0];
    float fy = pf[((size_t)b * nf + i) * 3 + 1];
    float fz = pf[((size_t)b * nf + i) * 3 + 2];
    float d0 = 3.4e38f, d1 = 3.4e38f, d2v = 3.4e38f;
    int   i0 = 0, i1 = 0, i2 = 0;
    for (int c = 0; c < nc; c++) {
        float dx = fx - PC[c * 3 + 0];
        float dy = fy - PC[c * 3 + 1];
        float dz = fz - PC[c * 3 + 2];
        float d  = dx * dx + dy * dy + dz * dz;
        if (d < d0)       { d2v = d1; i2 = i1; d1 = d0; i1 = i0; d0 = d; i0 = c; }
        else if (d < d1)  { d2v = d1; i2 = i1; d1 = d;  i1 = c; }
        else if (d < d2v) { d2v = d;  i2 = c; }
    }
    float w0 = 1.0f / fmaxf(d0, 1e-16f);
    float w1 = 1.0f / fmaxf(d1, 1e-16f);
    float w2 = 1.0f / fmaxf(d2v, 1e-16f);
    float s  = w0 + w1 + w2;
    size_t o = ((size_t)b * nf + i) * 3;
    ki[o + 0] = i0; ki[o + 1] = i1; ki[o + 2] = i2;
    kw[o + 0] = w0 / s; kw[o + 1] = w1 / s; kw[o + 2] = w2 / s;
}

// ---------------------------------------------------------------- weight f32 [K,N] -> padded TRANSPOSED bf16 [Np][Kp]
__global__ void k_cvt_w(const float* __restrict__ W, int K, int N, int Kp, int Np,
                        u16* __restrict__ Wt) {
    int t = blockIdx.x * blockDim.x + threadIdx.x;
    if (t >= Kp * Np) return;
    int n = t / Kp, k = t % Kp;
    Wt[t] = (k < K && n < N) ? f2bf(W[(size_t)k * N + n]) : (u16)0;
}

// ---------------------------------------------------------------- build SA input matrix [Mp x Kp] bf16
// row = j*32+k : [x[nb[j,k]] (C cols) | pos[nb]-q (3) | zero-pad]
__global__ void k_build_sa(const float* __restrict__ featF,  // non-null at level 0 (coords)
                           const u16* __restrict__ featB, int C,
                           const float* __restrict__ pos, const float* __restrict__ q,
                           const int* __restrict__ nb, const int* __restrict__ cnt,
                           int m, int Mp, int Kp, u16* __restrict__ A) {
    int row = blockIdx.x * blockDim.x + threadIdx.x;
    if (row >= Mp) return;
    u16* dst = A + (size_t)row * Kp;
    int valid = 0, s = 0, j = 0;
    if (row < m * 32) {
        j = row >> 5;
        int k = row & 31;
        if (k < cnt[j]) { valid = 1; s = nb[(size_t)j * 32 + k]; }
    }
    if (!valid) {
        for (int c = 0; c < Kp; c++) dst[c] = 0;
        return;
    }
    if (featF) {
        for (int c = 0; c < C; c++) dst[c] = f2bf(featF[(size_t)s * 3 + c]);
    } else {
        for (int c = 0; c < C; c++) dst[c] = featB[(size_t)s * C + c];
    }
    for (int c = 0; c < 3; c++)
        dst[C + c] = f2bf(pos[(size_t)s * 3 + c] - q[(size_t)j * 3 + c]);
    for (int c = C + 3; c < Kp; c++) dst[c] = 0;
}

// ---------------------------------------------------------------- masked max pool over K=32
__global__ void k_pool(const u16* __restrict__ H, int ldH, const int* __restrict__ cnt,
                       int m, int C, u16* __restrict__ out) {
    int t = blockIdx.x * blockDim.x + threadIdx.x;
    if (t >= m * C) return;
    int j = t / C, c = t % C;
    int cn = cnt[j];
    float best = 0.0f;
    if (cn > 0) {
        best = -3.4e38f;
        for (int k = 0; k < cn; k++)
            best = fmaxf(best, bf2f(H[((size_t)j * 32 + k) * ldH + c]));
    }
    out[(size_t)j * C + c] = f2bf(best);
}

// ---------------------------------------------------------------- build FP input [Mp x Kp] bf16
// row i : [ sum_k w*xc[idx_k] (Cc) | skip (Cs) | pf (3) | zero-pad ]
__global__ void k_build_fp(const u16* __restrict__ xc, int Cc,
                           const u16* __restrict__ skipB, const float* __restrict__ skipF, int Cs,
                           const float* __restrict__ pf,
                           const int* __restrict__ ki, const float* __restrict__ kw,
                           int nf, int Mp, int Kp, u16* __restrict__ A) {
    int row = blockIdx.x * blockDim.x + threadIdx.x;
    if (row >= Mp) return;
    u16* dst = A + (size_t)row * Kp;
    if (row >= nf) {
        for (int c = 0; c < Kp; c++) dst[c] = 0;
        return;
    }
    int   i0 = ki[(size_t)row * 3 + 0], i1 = ki[(size_t)row * 3 + 1], i2 = ki[(size_t)row * 3 + 2];
    float w0 = kw[(size_t)row * 3 + 0], w1 = kw[(size_t)row * 3 + 1], w2 = kw[(size_t)row * 3 + 2];
    for (int c = 0; c < Cc; c++) {
        float s = w0 * bf2f(xc[(size_t)i0 * Cc + c])
                + w1 * bf2f(xc[(size_t)i1 * Cc + c])
                + w2 * bf2f(xc[(size_t)i2 * Cc + c]);
        dst[c] = f2bf(s);
    }
    if (skipB) {
        for (int c = 0; c < Cs; c++) dst[Cc + c] = skipB[(size_t)row * Cs + c];
    } else {
        for (int c = 0; c < Cs; c++) dst[Cc + c] = f2bf(skipF[(size_t)row * 3 + c]);
    }
    for (int c = 0; c < 3; c++) dst[Cc + Cs + c] = f2bf(pf[(size_t)row * 3 + c]);
    for (int c = Cc + Cs + 3; c < Kp; c++) dst[c] = 0;
}

// ---------------------------------------------------------------- WMMA bf16 GEMM + bias + SiLU
// C[Mp x Np](bf16) = act( A[Mp x Kp](bf16) @ W (bf16, pre-transposed [Np][Kp]) + bias )
// 256 threads = 8 waves; 128x128 block tile; each wave a 32x64 tile (8 x v_wmma_f32_16x16x32_bf16 / K-step).
// Panels are DMA'd global->LDS with GLOBAL_LOAD_ASYNC_TO_LDS_B128 (ASYNCcnt protocol:
// issue -> s_wait_asynccnt 0 -> workgroup barrier), fragments are 2x ds_load_b128 each.
__global__ __launch_bounds__(256) void k_gemm(const u16* __restrict__ A, const u16* __restrict__ W,
                                              const float* __restrict__ bias, u16* __restrict__ C,
                                              int Mp, int Kp, int Np, int Nreal, int act) {
    __shared__ alignas(16) u16 lA[128 * 32];   // [row][k]
    __shared__ alignas(16) u16 lB[128 * 32];   // [col][k]  (weights already transposed)

    const int tid  = threadIdx.x;
    const int lane = tid & 31;
    const int wave = tid >> 5;
    const int wrow = (wave >> 1) * 32;   // 0,32,64,96
    const int wcol = (wave & 1) * 64;    // 0,64
    const int row0 = blockIdx.y * 128;
    const int col0 = blockIdx.x * 128;
    const int lrow = lane & 15;
    const int kh   = (lane >> 4) * 8;    // lanes 0-15: K 0-7/16-23; lanes 16-31: K 8-15/24-31

    const u32 lA0 = (u32)(size_t)lA;
    const u32 lB0 = (u32)(size_t)lB;
    const int sr  = tid >> 2;            // 0..63   (staged row/col, second trip +64)
    const int sg  = (tid & 3) * 8;       // 0,8,16,24 halves

    v8f z = {};
    v8f acc[2][4];
#pragma unroll
    for (int i = 0; i < 2; i++)
#pragma unroll
        for (int j = 0; j < 4; j++) acc[i][j] = z;

    for (int k0 = 0; k0 < Kp; k0 += 32) {
        // stage both panels: 4 unconditional async 16B DMAs per thread
        async_g2l_b128(lA0 + (u32)(sr * 32 + sg) * 2,
                       A + (size_t)(row0 + sr) * Kp + k0 + sg);
        async_g2l_b128(lA0 + (u32)((sr + 64) * 32 + sg) * 2,
                       A + (size_t)(row0 + sr + 64) * Kp + k0 + sg);
        async_g2l_b128(lB0 + (u32)(sr * 32 + sg) * 2,
                       W + (size_t)(col0 + sr) * Kp + k0 + sg);
        async_g2l_b128(lB0 + (u32)((sr + 64) * 32 + sg) * 2,
                       W + (size_t)(col0 + sr + 64) * Kp + k0 + sg);
        wait_async0();
        __syncthreads();

        v16bf afr[2], bfr[4];
#pragma unroll
        for (int i = 0; i < 2; i++) {
            const u16* p = lA + (wrow + i * 16 + lrow) * 32;
            v4u q0 = *(const v4u*)(p + kh);
            v4u q1 = *(const v4u*)(p + 16 + kh);
            v8u t;
            t[0] = q0[0]; t[1] = q0[1]; t[2] = q0[2]; t[3] = q0[3];
            t[4] = q1[0]; t[5] = q1[1]; t[6] = q1[2]; t[7] = q1[3];
            afr[i] = __builtin_bit_cast(v16bf, t);
        }
#pragma unroll
        for (int j = 0; j < 4; j++) {
            const u16* p = lB + (wcol + j * 16 + lrow) * 32;
            v4u q0 = *(const v4u*)(p + kh);
            v4u q1 = *(const v4u*)(p + 16 + kh);
            v8u t;
            t[0] = q0[0]; t[1] = q0[1]; t[2] = q0[2]; t[3] = q0[3];
            t[4] = q1[0]; t[5] = q1[1]; t[6] = q1[2]; t[7] = q1[3];
            bfr[j] = __builtin_bit_cast(v16bf, t);
        }
#pragma unroll
        for (int i = 0; i < 2; i++)
#pragma unroll
            for (int j = 0; j < 4; j++)
                acc[i][j] = __builtin_amdgcn_wmma_f32_16x16x32_bf16(
                    false, afr[i], false, bfr[j], (short)0, acc[i][j], false, false);
        __syncthreads();   // WAR: panels overwritten next iteration
    }

#pragma unroll
    for (int i = 0; i < 2; i++) {
#pragma unroll
        for (int j = 0; j < 4; j++) {
            int nn = col0 + wcol + j * 16 + lrow;
            float bv = (bias != nullptr && nn < Nreal) ? bias[nn] : 0.0f;
#pragma unroll
            for (int r = 0; r < 8; r++) {
                int mm = row0 + wrow + i * 16 + ((lane >> 4) * 8 + r);
                float v = acc[i][j][r] + bv;
                if (act) v = silu_f(v);
                C[(size_t)mm * Np + nn] = f2bf(v);
            }
        }
    }
}

// ---------------------------------------------------------------- residual: out[b,c,i] = fc3[i,c] + data[b,c,i]
__global__ void k_residual(const u16* __restrict__ fc3, int ld, const float* __restrict__ data,
                           float* __restrict__ out, int N, int b) {
    int i = blockIdx.x * blockDim.x + threadIdx.x;
    if (i >= N) return;
    for (int c = 0; c < 3; c++)
        out[(size_t)b * 3 * N + (size_t)c * N + i] =
            bf2f(fc3[(size_t)i * ld + c]) + data[(size_t)b * 3 * N + (size_t)c * N + i];
}

// ================================================================ host orchestration
extern "C" void kernel_launch(void* const* d_in, const int* in_sizes, int n_in,
                              void* d_out, int out_size, void* d_ws, size_t ws_size,
                              hipStream_t stream) {
    (void)n_in; (void)out_size; (void)ws_size;

    static const int   NPTS[5]   = {4096, 2868, 2008, 1406, 985};
    static const float R2[4]     = {0.01f, 0.09f, 0.25f, 0.49f};
    static const int   CL[5]     = {3, 64, 128, 256, 512};
    static const int   SA_CNT[4] = {3, 4, 4, 4};
    static const int   SA_K[4][4] = {{6,64,64,0},{67,64,64,64},{131,128,128,128},{259,256,256,256}};
    static const int   SA_N[4][4] = {{64,64,64,0},{64,64,64,128},{128,128,128,256},{256,256,256,512}};
    static const int   FP_CNT[4] = {4, 4, 2, 4};
    static const int   FP_K[4][4] = {{771,512,256,256},{387,256,256,256},{323,256,0,0},{262,256,128,128}};
    static const int   FP_N[4][4] = {{512,256,256,256},{256,256,256,256},{256,256,0,0},{256,128,128,128}};
    static const int   FC_K[4] = {128, 64, 32, 16};
    static const int   FC_N[4] = {64, 32, 16, 3};
    const int B = 4, N0 = 4096;

    auto cp = [](int x) { return (x + 127) & ~127; };   // pad to 128 (GEMM tile)

    // ---- input pointer tables (handle dict-insertion vs jax-sorted flattening)
    const float* saWp[4][4]; const float* saBp[4][4];
    const float* fpWp[4][4]; const float* fpBp[4][4];
    const float* fcWp[4];
    const float* data = (const float*)d_in[0];
    int p = 1;
    auto readSA = [&]() { for (int l = 0; l < 4; l++) for (int y = 0; y < SA_CNT[l]; y++) {
        saWp[l][y] = (const float*)d_in[p++]; saBp[l][y] = (const float*)d_in[p++]; } };
    auto readFP = [&]() { for (int l = 0; l < 4; l++) for (int y = 0; y < FP_CNT[l]; y++) {
        fpWp[l][y] = (const float*)d_in[p++]; fpBp[l][y] = (const float*)d_in[p++]; } };
    auto readFC = [&]() { for (int i = 0; i < 4; i++) fcWp[i] = (const float*)d_in[p++]; };
    if (in_sizes[1] == 128 * 64) { readFC(); readFP(); readSA(); }   // jax sorted-key order
    else                         { readSA(); readFP(); readFC(); }   // insertion order

    // ---- workspace bump allocator
    uint8_t* base = (uint8_t*)d_ws;
    size_t   off  = 0;
    auto alloc = [&](size_t bytes) -> void* {
        off = (off + 255) & ~(size_t)255;
        void* r = base + off;
        off += bytes;
        return r;
    };

    float* posL[5];
    for (int l = 0; l < 5; l++) posL[l] = (float*)alloc((size_t)B * NPTS[l] * 3 * 4);
    int* idxL[4]; int* nbL[4]; int* cntL[4];
    for (int l = 0; l < 4; l++) {
        idxL[l] = (int*)alloc((size_t)B * NPTS[l + 1] * 4);
        nbL[l]  = (int*)alloc((size_t)B * NPTS[l + 1] * 32 * 4);
        cntL[l] = (int*)alloc((size_t)B * NPTS[l + 1] * 4);
    }
    int* knnI[4]; float* knnW[4];
    for (int f = 0; f < 4; f++) {
        int nf = NPTS[3 - f];
        knnI[f] = (int*)alloc((size_t)B * nf * 3 * 4);
        knnW[f] = (float*)alloc((size_t)B * nf * 3 * 4);
    }
    u16* feats[5] = {nullptr, nullptr, nullptr, nullptr, nullptr};
    for (int l = 1; l <= 4; l++) feats[l] = (u16*)alloc((size_t)NPTS[l] * CL[l] * 2);
    u16* fpOut[4];
    for (int f = 0; f < 4; f++)
        fpOut[f] = (u16*)alloc((size_t)cp(NPTS[3 - f]) * cp(FP_N[f][FP_CNT[f] - 1]) * 2);

    // padded, transposed bf16 weights
    u16* saW[4][4]; u16* fpW[4][4]; u16* fcW[4];
    for (int l = 0; l < 4; l++) for (int y = 0; y < SA_CNT[l]; y++)
        saW[l][y] = (u16*)alloc((size_t)cp(SA_K[l][y]) * cp(SA_N[l][y]) * 2);
    for (int l = 0; l < 4; l++) for (int y = 0; y < FP_CNT[l]; y++)
        fpW[l][y] = (u16*)alloc((size_t)cp(FP_K[l][y]) * cp(FP_N[l][y]) * 2);
    for (int i = 0; i < 4; i++)
        fcW[i] = (u16*)alloc((size_t)cp(FC_K[i]) * cp(FC_N[i]) * 2);

    const size_t ACT_ELEMS = (size_t)31616 * 512;   // largest padded activation (SA3 hidden)
    u16* actA = (u16*)alloc(ACT_ELEMS * 2);
    u16* actB = (u16*)alloc(ACT_ELEMS * 2);

    auto cvt = [&](const float* W, int K, int Nn, u16* Wp) {
        int Kp = cp(K), Np = cp(Nn);
        int tot = Kp * Np;
        k_cvt_w<<<(tot + 255) / 256, 256, 0, stream>>>(W, K, Nn, Kp, Np, Wp);
    };
    for (int l = 0; l < 4; l++) for (int y = 0; y < SA_CNT[l]; y++) cvt(saWp[l][y], SA_K[l][y], SA_N[l][y], saW[l][y]);
    for (int l = 0; l < 4; l++) for (int y = 0; y < FP_CNT[l]; y++) cvt(fpWp[l][y], FP_K[l][y], FP_N[l][y], fpW[l][y]);
    for (int i = 0; i < 4; i++) cvt(fcWp[i], FC_K[i], FC_N[i], fcW[i]);

    // ---- geometry (all batches)
    k_pos0<<<dim3((N0 + 255) / 256, B), 256, 0, stream>>>(data, posL[0], N0);
    for (int l = 0; l < 4; l++) {
        int n = NPTS[l], m = NPTS[l + 1];
        k_fps<<<dim3(B), dim3(1024), 0, stream>>>(posL[l], n, m, idxL[l]);
        k_gather_pos<<<dim3((m + 255) / 256, B), 256, 0, stream>>>(posL[l], idxL[l], posL[l + 1], n, m);
        k_ball<<<dim3((m + 255) / 256, B), 256, 0, stream>>>(posL[l], posL[l + 1], n, m, R2[l], nbL[l], cntL[l]);
    }
    for (int f = 0; f < 4; f++) {
        int nc = NPTS[4 - f], nf = NPTS[3 - f];
        k_knn<<<dim3((nf + 255) / 256, B), 256, 0, stream>>>(posL[4 - f], posL[3 - f], nc, nf, knnI[f], knnW[f]);
    }

    auto gemm = [&](const u16* Ain, int Mp, int Kp, const u16* Wp, const float* bias,
                    int Nn, u16* Cout, int act) {
        dim3 g(cp(Nn) / 128, Mp / 128), blk(256);
        k_gemm<<<g, blk, 0, stream>>>(Ain, Wp, bias, Cout, Mp, Kp, cp(Nn), Nn, act);
    };

    // ---- MLP phase, batch-sequential (reuses act buffers)
    for (int b = 0; b < B; b++) {
        // SA levels
        for (int l = 0; l < 4; l++) {
            int n = NPTS[l], m = NPTS[l + 1];
            int Mp  = cp(m * 32);
            int Kp0 = cp(SA_K[l][0]);
            const float* fF = (l == 0) ? (posL[0] + (size_t)b * n * 3) : nullptr;
            const u16*   fB = (l == 0) ? nullptr : feats[l];
            k_build_sa<<<(Mp + 255) / 256, 256, 0, stream>>>(
                fF, fB, CL[l], posL[l] + (size_t)b * n * 3, posL[l + 1] + (size_t)b * m * 3,
                nbL[l] + (size_t)b * m * 32, cntL[l] + (size_t)b * m, m, Mp, Kp0, actA);
            const u16* cur = actA; int curKp = Kp0;
            for (int y = 0; y < SA_CNT[l]; y++) {
                u16* nxt = (cur == actA) ? actB : actA;
                gemm(cur, Mp, curKp, saW[l][y], saBp[l][y], SA_N[l][y], nxt, 1);
                cur = nxt; curKp = cp(SA_N[l][y]);
            }
            int C = SA_N[l][SA_CNT[l] - 1];
            k_pool<<<((m * C) + 255) / 256, 256, 0, stream>>>(
                cur, curKp, cntL[l] + (size_t)b * m, m, C, feats[l + 1]);
        }
        // FP levels
        const u16* xc = feats[4]; int Cc = 512;
        for (int f = 0; f < 4; f++) {
            int lf = 3 - f, nf = NPTS[lf], Mp = cp(nf);
            int Kp0 = cp(FP_K[f][0]);
            const u16*   skB = (lf == 0) ? nullptr : feats[lf];
            const float* skF = (lf == 0) ? (posL[0] + (size_t)b * N0 * 3) : nullptr;
            int Cs = (lf == 0) ? 3 : CL[lf];
            k_build_fp<<<(Mp + 255) / 256, 256, 0, stream>>>(
                xc, Cc, skB, skF, Cs, posL[lf] + (size_t)b * nf * 3,
                knnI[f] + (size_t)b * nf * 3, knnW[f] + (size_t)b * nf * 3, nf, Mp, Kp0, actA);
            const u16* cur = actA; int curKp = Kp0;
            for (int y = 0; y < FP_CNT[f]; y++) {
                u16* nxt = (y == FP_CNT[f] - 1) ? fpOut[f] : ((cur == actA) ? actB : actA);
                gemm(cur, Mp, curKp, fpW[f][y], fpBp[f][y], FP_N[f][y], nxt, 1);
                cur = nxt; curKp = cp(FP_N[f][y]);
            }
            xc = fpOut[f]; Cc = cp(FP_N[f][FP_CNT[f] - 1]);
        }
        // FC head: silu only after first layer; no biases
        gemm(fpOut[3], 4096, 128, fcW[0], nullptr, FC_N[0], actA, 1);
        gemm(actA, 4096, cp(FC_N[0]), fcW[1], nullptr, FC_N[1], actB, 0);
        gemm(actB, 4096, cp(FC_N[1]), fcW[2], nullptr, FC_N[2], actA, 0);
        gemm(actA, 4096, cp(FC_N[2]), fcW[3], nullptr, FC_N[3], actB, 0);
        k_residual<<<(N0 + 255) / 256, 256, 0, stream>>>(actB, cp(FC_N[3]), data, (float*)d_out, N0, b);
    }
}
